// MultiResolutionMerger_80264348828011
// MI455X (gfx1250) — matrix operand order repaired and checked by reading, not compiled, not run on hardware
//
#include <hip/hip_runtime.h>

// ---------------------------------------------------------------------------
// MultiResolutionMerger for MI455X (gfx1250, wave32, WMMA + async-to-LDS)
//
// Phase 1 (gate_kernel): per-(b,m) gating weights, normalized over c,
//   stored f16 in workspace [B][128][KPAD=320] (K zero-padded).
// Phase 2 (gemm_kernel): out[b](128x4000) = W[b](128x306) @ x[b](306x4000)
//   with v_wmma_f32_16x16x32_f16. x tiles staged into LDS once per block
//   via inline-asm global_load_async_to_lds_b128 (double buffered,
//   ASYNCcnt), shared by all 8 waves -> 8x fewer VMEM fetches of x.
// ---------------------------------------------------------------------------

typedef __attribute__((ext_vector_type(16))) _Float16 v16h;
typedef __attribute__((ext_vector_type(8)))  _Float16 v8h;
typedef __attribute__((ext_vector_type(8)))  float    v8f;

#define B_   16
#define C_   306
#define T_   4000
#define M_   128
#define KPAD 320                 // C padded to multiple of 32
#define NKT  (KPAD / 32)         // 10 K-tiles

// ---------------------------------------------------------------------------
// Phase 1: gating MLP + channel normalization. grid = B_*M_, block = 320.
// ---------------------------------------------------------------------------
__global__ __launch_bounds__(320)
void gate_kernel(const float* __restrict__ pos,   // [B][C][2]
                 const float* __restrict__ tp,    // [M][2]
                 const float* __restrict__ w1,    // [3][32]
                 const float* __restrict__ b1,    // [32]
                 const float* __restrict__ w2,    // [32]
                 const float* __restrict__ b2,    // [1]
                 _Float16* __restrict__ Wf)       // [B][128][KPAD]
{
    __shared__ float sW1[96];
    __shared__ float sB1[32];
    __shared__ float sW2[32];
    __shared__ float sB2;
    __shared__ float red[512];

    const int b = blockIdx.x >> 7;
    const int m = blockIdx.x & 127;
    const int c = threadIdx.x;

    if (threadIdx.x < 96) sW1[threadIdx.x] = w1[threadIdx.x];
    if (threadIdx.x < 32) { sB1[threadIdx.x] = b1[threadIdx.x];
                            sW2[threadIdx.x] = w2[threadIdx.x]; }
    if (threadIdx.x == 0) sB2 = b2[0];
    if (threadIdx.x < 192) red[320 + threadIdx.x] = 0.0f;   // pad to 512
    __syncthreads();

    float wgt = 0.0f;
    if (c < C_) {
        const float px = pos[((size_t)b * C_ + c) * 2 + 0];
        const float py = pos[((size_t)b * C_ + c) * 2 + 1];
        const float tx = tp[m * 2 + 0];
        const float ty = tp[m * 2 + 1];
        const float dx = px - tx, dy = py - ty;
        const float d2 = dx * dx + dy * dy;
        const float s0 = expf(-d2 * 50.0f);    // 1/(2*0.1^2)
        const float s1 = expf(-d2 * 12.5f);    // 1/(2*0.2^2)
        const float s2 = expf(-d2 * 3.125f);   // 1/(2*0.4^2)
        float acc = sB2;
#pragma unroll
        for (int j = 0; j < 32; ++j) {
            float h = fmaf(s0, sW1[j],
                      fmaf(s1, sW1[32 + j],
                      fmaf(s2, sW1[64 + j], sB1[j])));
            h = fmaxf(h, 0.0f);
            acc = fmaf(h, sW2[j], acc);
        }
        wgt = acc;
    }

    red[threadIdx.x] = wgt;
    __syncthreads();
#pragma unroll
    for (int off = 256; off >= 1; off >>= 1) {
        if (threadIdx.x < off) red[threadIdx.x] += red[threadIdx.x + off];
        __syncthreads();
    }
    const float rnorm = 1.0f / (red[0] + 1e-8f);

    const _Float16 o = (c < C_) ? (_Float16)(wgt * rnorm) : (_Float16)0.0f;
    Wf[((size_t)(b * M_ + m)) * KPAD + c] = o;
}

// ---------------------------------------------------------------------------
// Async copy helpers (gfx1250). A generic pointer to __shared__ memory is
// (aperture<<32)|lds_offset, so its low 32 bits are exactly the wave-relative
// LDS byte address that global_load_async_to_lds_* wants in VDST.
// ---------------------------------------------------------------------------
__device__ __forceinline__ void async_copy_b128(const float* gsrc, float* lds_dst) {
    const unsigned           ldsa = (unsigned)(unsigned long long)(uintptr_t)lds_dst;
    const unsigned long long ga   = (unsigned long long)(uintptr_t)gsrc;
    asm volatile("global_load_async_to_lds_b128 %0, %1, off"
                 :: "v"(ldsa), "v"(ga)
                 : "memory");
}

__device__ __forceinline__ void async_wait0() {
#if defined(__has_builtin) && __has_builtin(__builtin_amdgcn_s_wait_asynccnt)
    __builtin_amdgcn_s_wait_asynccnt(0);
#else
    asm volatile("s_wait_asynccnt 0x0" ::: "memory");
#endif
}

// ---------------------------------------------------------------------------
// Phase 2 LDS tile layout: 32 K-rows x 32 T-cols of f32.
// Rows 0..15 at rowOff = r*36; rows 16..31 at 592 + (r-16)*36.
//   - stride 36 floats keeps every b128 destination 16B-aligned,
//   - the +592 (%64 banks = +16) half-split puts the two half-wave read
//     windows in disjoint bank ranges (no LDS bank conflicts).
// ---------------------------------------------------------------------------
#define ROWSTR   36
#define HALFOFF  592
#define TILEF32  1184            // floats per buffer

__device__ __forceinline__ int row_off(int r) {
    return (r & 15) * ROWSTR + (r >> 4) * HALFOFF;
}

__global__ __launch_bounds__(256)
void gemm_kernel(const float* __restrict__ x,     // [B][C][T]
                 const _Float16* __restrict__ Wf, // [B][128][KPAD]
                 float* __restrict__ out)         // [B][128][T]
{
    __shared__ __align__(16) float tile[2][TILEF32];

    const int b     = blockIdx.y;
    const int tbase = blockIdx.x << 5;            // 32-wide T tile
    const int wave  = threadIdx.x >> 5;
    const int lane  = threadIdx.x & 31;
    const int n     = lane & 15;
    const int half  = lane >> 4;
    const int mbase = wave << 4;

    const float*    xb   = x  + (size_t)b * C_ * T_;
    const _Float16* wrow = Wf + ((size_t)(b * M_ + mbase + n)) * KPAD;

    // ---- cooperative stage of one 32x32 f32 x-tile into an LDS buffer ----
    const int sr = threadIdx.x >> 3;              // 0..31 : K row in tile
    const int sc = (threadIdx.x & 7) << 2;        // 0,4,..,28 : T col (x4)
    auto stage = [&](int kt, int buf) {
        const int crow = kt * 32 + sr;
        float* dst = &tile[buf][row_off(sr) + sc];
        if (crow < C_) {
            async_copy_b128(xb + (size_t)crow * T_ + tbase + sc, dst);
        } else {
            dst[0] = 0.0f; dst[1] = 0.0f; dst[2] = 0.0f; dst[3] = 0.0f;
        }
    };
    auto stage_wait = [&]() {
        async_wait0();
        __syncthreads();
    };

    v8f acc0 = {};
    v8f acc1 = {};

    stage(0, 0);
    stage_wait();

    for (int kt = 0; kt < NKT; ++kt) {
        const int buf = kt & 1;
        if (kt + 1 < NKT) stage(kt + 1, buf ^ 1);

        // ---- A fragment: two 16B global loads from padded f16 weights ----
        //   element e holds K = 32*kt + 16*(e>>3) + 8*half + (e&7)
        const _Float16* ap = wrow + kt * 32 + 8 * half;
        const v8h alo = *(const v8h*)(ap);
        const v8h ahi = *(const v8h*)(ap + 16);
        v16h afrag;
#pragma unroll
        for (int e = 0; e < 8; ++e) { afrag[e] = alo[e]; afrag[8 + e] = ahi[e]; }

        // ---- B fragments from LDS (cvt f32->f16); element e: K row 16*(e>>3)+8*half+(e&7)
        const float* bp = &tile[buf][half * 8 * ROWSTR + n];
        v16h bfrag0, bfrag1;
#pragma unroll
        for (int e = 0; e < 16; ++e) {
            bfrag0[e] = (_Float16)bp[ROWSTR * (e & 7) + HALFOFF * (e >> 3)];        // T cols [0,16)
            bfrag1[e] = (_Float16)bp[ROWSTR * (e & 7) + HALFOFF * (e >> 3) + 16];   // T cols [16,32)
        }

        acc0 = __builtin_amdgcn_wmma_f32_16x16x32_f16(
                   false, afrag, false, bfrag0, (short)0, acc0, false, false);
        acc1 = __builtin_amdgcn_wmma_f32_16x16x32_f16(
                   false, afrag, false, bfrag1, (short)0, acc1, false, false);

        if (kt + 1 < NKT) stage_wait();
    }

    // ---- store D: VGPR i -> row mbase + i + 8*half, col tbase + nt*16 + n
    float* op = out + ((size_t)(b * M_ + mbase + 8 * half)) * T_ + tbase + n;
#pragma unroll
    for (int i = 0; i < 8; ++i) {
        op[(size_t)i * T_]      = acc0[i];
        op[(size_t)i * T_ + 16] = acc1[i];
    }
}

// ---------------------------------------------------------------------------
extern "C" void kernel_launch(void* const* d_in, const int* in_sizes, int n_in,
                              void* d_out, int out_size, void* d_ws, size_t ws_size,
                              hipStream_t stream) {
    const float* x   = (const float*)d_in[0];
    const float* pos = (const float*)d_in[1];
    const float* tp  = (const float*)d_in[2];
    const float* w1  = (const float*)d_in[3];
    const float* b1  = (const float*)d_in[4];
    const float* w2  = (const float*)d_in[5];
    const float* b2  = (const float*)d_in[6];
    (void)in_sizes; (void)n_in; (void)out_size; (void)ws_size;

    _Float16* Wf = (_Float16*)d_ws;   // 16*128*320 f16 = 1.25 MB scratch
    float* out = (float*)d_out;

    gate_kernel<<<dim3(B_ * M_), dim3(320), 0, stream>>>(pos, tp, w1, b1, w2, b2, Wf);
    gemm_kernel<<<dim3(T_ / 32, B_), dim3(256), 0, stream>>>(x, Wf, out);
}
